// TopKSAE_26499948216784
// MI455X (gfx1250) — matrix operand order repaired and checked by reading, not compiled, not run on hardware
//
#include <hip/hip_runtime.h>

// ---------------------------------------------------------------- constants
#define N_ROWS   8192
#define D_IN     2048
#define D_HIDDEN 32768
#define TOPK     64

#define M_TILE   64      // rows per block in encoder
#define N_CHUNK  256     // hidden cols per chunk (8 waves x 2 col-tiles)
#define K_CHUNK  32      // bf16 WMMA K
#define KCHUNKS  (D_IN / K_CHUNK)      // 64
#define HSPLIT   8       // hidden-dim split across blocks
#define HID_PER  (D_HIDDEN / HSPLIT)   // 4096
#define NCHUNKS  (HID_PER / N_CHUNK)   // 16

#define AS_BYTES (M_TILE * K_CHUNK * 2)    // 4096
#define BS_BYTES (N_CHUNK * K_CHUNK * 2)   // 16384

typedef __attribute__((ext_vector_type(16))) __bf16 bf16x16;
typedef __attribute__((ext_vector_type(8)))  float  f32x8;

union Frag32 {            // 32 bytes: one WMMA bf16 A/B operand per lane
    uint4   u[2];
    bf16x16 v;
};

__device__ __forceinline__ unsigned short f32_to_bf16(float f) {
    unsigned int u = __float_as_uint(f);
    unsigned int r = u + 0x7FFFu + ((u >> 16) & 1u);   // round-to-nearest-even
    return (unsigned short)(r >> 16);
}
__device__ __forceinline__ float bf16_to_f32(unsigned short s) {
    return __uint_as_float(((unsigned int)s) << 16);
}

// CDNA5 async global->LDS copy (ASYNCcnt-tracked), 16 bytes per lane.
__device__ __forceinline__ void async_b128(unsigned lds_addr, const void* gptr) {
    asm volatile("global_load_async_to_lds_b128 %0, %1, off"
                 :: "v"(lds_addr), "v"((unsigned long long)(uintptr_t)gptr)
                 : "memory");
}
#define S_WAIT_ASYNC(n) asm volatile("s_wait_asynccnt " #n ::: "memory")

// ------------------------------------------------- kernel 1a: pack (x - bias)
__global__ __launch_bounds__(256)
void pack_x_kernel(const float* __restrict__ x, const float* __restrict__ dbias,
                   unsigned short* __restrict__ out, size_t n) {
    size_t stride = (size_t)gridDim.x * blockDim.x;
    for (size_t i = (size_t)blockIdx.x * blockDim.x + threadIdx.x; i < n; i += stride)
        out[i] = f32_to_bf16(x[i] - dbias[i & (D_IN - 1)]);   // NORM_CONSTANT == 1.0
}

// ------------------------------------------------- kernel 1b: pack a matrix
__global__ __launch_bounds__(256)
void pack_bf16_kernel(const float* __restrict__ in, unsigned short* __restrict__ out,
                      size_t n) {
    size_t stride = (size_t)gridDim.x * blockDim.x;
    for (size_t i = (size_t)blockIdx.x * blockDim.x + threadIdx.x; i < n; i += stride)
        out[i] = f32_to_bf16(in[i]);
}

// ---------------------------------- kernel 2: WMMA GEMM + ReLU + slice top-64
// grid = (N_ROWS/M_TILE, HSPLIT), block = 256 (8 waves)
// Double-buffered async global->LDS pipeline over K chunks:
//   5 async_b128 per thread per chunk (A: 1, B: 4); wait asynccnt<=5 keeps
//   exactly one chunk in flight while WMMAs consume the other buffer.
__global__ __launch_bounds__(256)
void encode_topk_kernel(const unsigned short* __restrict__ xb,
                        const unsigned short* __restrict__ wb,
                        float* __restrict__ pvals, int* __restrict__ pidx) {
    extern __shared__ char smem[];
    unsigned short* As   = (unsigned short*)smem;                    // 2 x [64][32]
    unsigned short* Bs   = (unsigned short*)(smem + 2 * AS_BYTES);   // 2 x [256][32]
    float* preC          = (float*)(smem + 2 * AS_BYTES + 2 * BS_BYTES); // [64][256]
    float* topV          = preC + M_TILE * N_CHUNK;                  // [64][64]
    int*   topI          = (int*)(topV + M_TILE * TOPK);             // [64][64]
    float* rowMin        = (float*)(topI + M_TILE * TOPK);           // [64]
    int*   rowMinP       = (int*)(rowMin + M_TILE);                  // [64]

    const int tid  = threadIdx.x;
    const int lane = tid & 31;
    const int wave = tid >> 5;
    const int half = lane >> 4;      // which 16-lane half of the wave
    const int ln   = lane & 15;
    const int row0 = blockIdx.x * M_TILE;
    const int hid0 = blockIdx.y * HID_PER;

    const unsigned asBase = (unsigned)(uintptr_t)As;   // LDS byte offsets
    const unsigned bsBase = (unsigned)(uintptr_t)Bs;

    // per-thread staging coordinates (16B units)
    const int ar = tid >> 2, ac = tid & 3;            // A: 64 rows x 4 units

    for (int i = tid; i < M_TILE * TOPK; i += 256) { topV[i] = -1.0f; topI[i] = 0; }
    if (tid < M_TILE) { rowMin[tid] = -1.0f; rowMinP[tid] = 0; }
    __syncthreads();

    for (int nc = 0; nc < NCHUNKS; ++nc) {
        const int col0 = hid0 + nc * N_CHUNK;

        f32x8 acc0[4], acc1[4];
        for (int r = 0; r < 4; ++r)
            for (int e = 0; e < 8; ++e) { acc0[r][e] = 0.0f; acc1[r][e] = 0.0f; }

        // ---- prologue: issue K-chunks 0 and 1 into buffers 0 and 1
        for (int p = 0; p < 2; ++p) {
            const int kb = p * K_CHUNK;
            const unsigned aB = asBase + p * AS_BYTES;
            const unsigned bB = bsBase + p * BS_BYTES;
            async_b128(aB + ar * 64 + ac * 16,
                       xb + (size_t)(row0 + ar) * D_IN + kb + ac * 8);
#pragma unroll
            for (int i = 0; i < 4; ++i) {
                const int u = tid + i * 256;          // B: 256 rows x 4 units
                const int br = u >> 2, bc = u & 3;
                async_b128(bB + br * 64 + bc * 16,
                           wb + (size_t)(col0 + br) * D_IN + kb + bc * 8);
            }
        }

        for (int kc = 0; kc < KCHUNKS; ++kc) {
            const int buf = kc & 1;
            // wait: at most the *next* chunk's 5 async ops outstanding
            if (kc + 1 < KCHUNKS) { S_WAIT_ASYNC(5); } else { S_WAIT_ASYNC(0); }
            __syncthreads();                           // buf fully written, visible

            const unsigned short* Ab = (const unsigned short*)(smem + buf * AS_BYTES);
            const unsigned short* Bb = (const unsigned short*)(smem + 2 * AS_BYTES + buf * BS_BYTES);

            // B fragments: two col-tiles per wave; K = half*16 + 0..15 contiguous
            Frag32 b0, b1;
            {
                const unsigned short* p0 = Bb + (wave * 32 + ln) * K_CHUNK + half * 16;
                b0.u[0] = ((const uint4*)p0)[0];
                b0.u[1] = ((const uint4*)p0)[1];
                const unsigned short* p1 = p0 + 16 * K_CHUNK;
                b1.u[0] = ((const uint4*)p1)[0];
                b1.u[1] = ((const uint4*)p1)[1];
            }
#pragma unroll
            for (int r = 0; r < 4; ++r) {
                // A fragment: row m = r*16+ln; K = half*16 + 0..15 contiguous
                Frag32 a;
                const unsigned short* p = Ab + (r * 16 + ln) * K_CHUNK + half * 16;
                a.u[0] = ((const uint4*)p)[0];
                a.u[1] = ((const uint4*)(p + 8))[0];
                acc0[r] = __builtin_amdgcn_wmma_f32_16x16x32_bf16(
                    false, a.v, false, b0.v, (short)0, acc0[r], false, false);
                acc1[r] = __builtin_amdgcn_wmma_f32_16x16x32_bf16(
                    false, a.v, false, b1.v, (short)0, acc1[r], false, false);
            }
            __syncthreads();                           // all waves done reading buf

            // ---- issue chunk kc+2 into the buffer we just finished reading
            if (kc + 2 < KCHUNKS) {
                const int kb = (kc + 2) * K_CHUNK;
                const unsigned aB = asBase + buf * AS_BYTES;
                const unsigned bB = bsBase + buf * BS_BYTES;
                async_b128(aB + ar * 64 + ac * 16,
                           xb + (size_t)(row0 + ar) * D_IN + kb + ac * 8);
#pragma unroll
                for (int i = 0; i < 4; ++i) {
                    const int u = tid + i * 256;
                    const int br = u >> 2, bc = u & 3;
                    async_b128(bB + br * 64 + bc * 16,
                               wb + (size_t)(col0 + br) * D_IN + kb + bc * 8);
                }
            }
        }

        // ReLU + dump C tiles to LDS (C layout: elem e -> m = half*8+e, n = ln)
#pragma unroll
        for (int r = 0; r < 4; ++r)
#pragma unroll
            for (int e = 0; e < 8; ++e) {
                const int m = r * 16 + half * 8 + e;
                const int n0 = wave * 32 + ln;
                float v = acc0[r][e];
                preC[m * N_CHUNK + n0] = v > 0.0f ? v : 0.0f;
                v = acc1[r][e];
                preC[m * N_CHUNK + n0 + 16] = v > 0.0f ? v : 0.0f;
            }
        __syncthreads();

        // per-row running top-64 (one thread per row; threshold rejects most)
        if (tid < M_TILE) {
            float mn = rowMin[tid];
            int   mp = rowMinP[tid];
            float* tv = topV + tid * TOPK;
            int*   ti = topI + tid * TOPK;
            const float* pr = preC + tid * N_CHUNK;
            for (int c = 0; c < N_CHUNK; ++c) {
                const float v = pr[c];
                if (v > mn) {
                    tv[mp] = v; ti[mp] = col0 + c;
                    mn = tv[0]; mp = 0;
                    for (int j = 1; j < TOPK; ++j) {
                        const float t = tv[j];
                        if (t < mn) { mn = t; mp = j; }
                    }
                }
            }
            rowMin[tid] = mn; rowMinP[tid] = mp;
        }
        __syncthreads();
    }

    // write partial top-64 lists: [row][split][64]
    for (int i = tid; i < M_TILE * TOPK; i += 256) {
        const int r = i >> 6, j = i & 63;
        const size_t o = ((size_t)(row0 + r) * HSPLIT + blockIdx.y) * TOPK + j;
        pvals[o] = topV[i];
        pidx[o]  = topI[i];
    }
}

// ------------------- kernel 3: merge 8x64 partials -> top-64, then decode row
// grid = N_ROWS, block = 256
__global__ __launch_bounds__(256)
void merge_decode_kernel(const float* __restrict__ pvals, const int* __restrict__ pidx,
                         const unsigned short* __restrict__ db,
                         const float* __restrict__ dbias,
                         float* __restrict__ xhat, float* __restrict__ ovals,
                         int* __restrict__ oidx) {
    __shared__ float cv[HSPLIT * TOPK];
    __shared__ int   ci[HSPLIT * TOPK];
    __shared__ float rv[256];
    __shared__ int   rp[256];
    __shared__ float sv[TOPK];
    __shared__ int   si[TOPK];

    const int row = blockIdx.x;
    const int tid = threadIdx.x;

    for (int i = tid; i < HSPLIT * TOPK; i += 256) {
        cv[i] = pvals[(size_t)row * (HSPLIT * TOPK) + i];
        ci[i] = pidx[(size_t)row * (HSPLIT * TOPK) + i];
    }
    __syncthreads();

    // 64 rounds of parallel arg-max over 512 candidates -> descending order
    for (int k = 0; k < TOPK; ++k) {
        float v0 = cv[tid], v1 = cv[tid + 256];
        float bv; int bp;
        if (v1 > v0) { bv = v1; bp = tid + 256; } else { bv = v0; bp = tid; }
        rv[tid] = bv; rp[tid] = bp;
        __syncthreads();
        for (int s = 128; s > 0; s >>= 1) {
            if (tid < s && rv[tid + s] > rv[tid]) { rv[tid] = rv[tid + s]; rp[tid] = rp[tid + s]; }
            __syncthreads();
        }
        if (tid == 0) {
            const int p = rp[0];
            sv[k] = rv[0]; si[k] = ci[p];
            cv[p] = -2.0f;                       // remove from candidate pool
        }
        __syncthreads();
    }

    if (tid < TOPK) {
        ovals[(size_t)row * TOPK + tid] = sv[tid];
        oidx [(size_t)row * TOPK + tid] = si[tid];
    }

    // decode: x_hat[row] = bias + sum_k v_k * decoder_bf16[idx_k]
    float acc[D_IN / 256];
#pragma unroll
    for (int i = 0; i < D_IN / 256; ++i) acc[i] = dbias[tid + i * 256];
    for (int k = 0; k < TOPK; ++k) {
        const float v = sv[k];
        const unsigned short* dr = db + (size_t)si[k] * D_IN;
#pragma unroll
        for (int i = 0; i < D_IN / 256; ++i)
            acc[i] += v * bf16_to_f32(dr[tid + i * 256]);
    }
#pragma unroll
    for (int i = 0; i < D_IN / 256; ++i)
        xhat[(size_t)row * D_IN + tid + i * 256] = acc[i];
}

// ---------------------------------------------------------------- launcher
extern "C" void kernel_launch(void* const* d_in, const int* in_sizes, int n_in,
                              void* d_out, int out_size, void* d_ws, size_t ws_size,
                              hipStream_t stream) {
    const float* x     = (const float*)d_in[0];   // [8192, 2048]
    const float* W     = (const float*)d_in[1];   // [32768, 2048]
    const float* dec   = (const float*)d_in[2];   // [32768, 2048]
    const float* dbias = (const float*)d_in[3];   // [2048]

    const size_t xn = (size_t)N_ROWS * D_IN;          // 16,777,216
    const size_t wn = (size_t)D_HIDDEN * D_IN;        // 67,108,864

    char* ws = (char*)d_ws;
    unsigned short* xb = (unsigned short*)ws;                       // 32 MB
    unsigned short* wb = (unsigned short*)(ws + xn * 2);            // 128 MB
    unsigned short* db = (unsigned short*)(ws + xn * 2 + wn * 2);   // 128 MB
    float* pv = (float*)(ws + xn * 2 + 2 * wn * 2);                 // 16 MB
    int*   pi = (int*)  ((char*)pv + (size_t)N_ROWS * HSPLIT * TOPK * 4); // 16 MB

    float* xhat  = (float*)d_out;
    float* ovals = xhat + xn;
    int*   oidx  = (int*)(ovals + (size_t)N_ROWS * TOPK);

    pack_x_kernel   <<<2048, 256, 0, stream>>>(x, dbias, xb, xn);
    pack_bf16_kernel<<<8192, 256, 0, stream>>>(W,   wb, wn);
    pack_bf16_kernel<<<8192, 256, 0, stream>>>(dec, db, wn);

    const size_t smem =
        2 * AS_BYTES +                      // As double buffer
        2 * BS_BYTES +                      // Bs double buffer
        (size_t)M_TILE * N_CHUNK * 4 +      // preC
        (size_t)M_TILE * TOPK * 4 * 2 +     // topV + topI
        (size_t)M_TILE * 8;                 // rowMin + rowMinP
    dim3 grid2(N_ROWS / M_TILE, HSPLIT);
    encode_topk_kernel<<<grid2, 256, smem, stream>>>(xb, wb, pv, pi);

    merge_decode_kernel<<<N_ROWS, 256, 0, stream>>>(pv, pi, db, dbias,
                                                    xhat, ovals, oidx);
}
